// TransformerLayer_64622077935849
// MI455X (gfx1250) — matrix operand order repaired and checked by reading, not compiled, not run on hardware
//
#include <hip/hip_runtime.h>
#include <hip/hip_bf16.h>

typedef _Float16 half_t;
typedef __attribute__((ext_vector_type(16))) _Float16 v16h;
typedef __attribute__((ext_vector_type(8)))  float    v8f;

#define N_TOK 4096
#define D_MOD 128
#define K_HD  64
#define H_N   4
#define F_FFN 256
#define HV    256   // H * V

// ---------------------------------------------------------------------------
// Async-to-LDS support (CDNA5 §08): ASYNCcnt-tracked DMA, in-order completion.
// ---------------------------------------------------------------------------
#if defined(__has_builtin)
#if __has_builtin(__builtin_amdgcn_s_wait_asynccnt)
#define WAIT_ASYNC(n) __builtin_amdgcn_s_wait_asynccnt(n)
#endif
#endif
#ifndef WAIT_ASYNC
#define WAIT_ASYNC(n) asm volatile("s_wait_asynccnt %0" ::"n"(n) : "memory")
#endif

// Low 32 bits of a generic pointer into the LDS aperture are the LDS byte
// address (ISA 10.2).
static __device__ __forceinline__ unsigned lds_addr(const void* p) {
  return (unsigned)(uintptr_t)p;
}

// GVS-mode async copy: LDS[vdst] <= MEM[saddr + vaddr], 8 bytes per lane.
static __device__ __forceinline__ void async_copy_b64(unsigned lds_off,
                                                      const void* sbase,
                                                      unsigned byte_off) {
  asm volatile("global_load_async_to_lds_b64 %0, %1, %2"
               :
               : "v"(lds_off), "v"(byte_off),
                 "s"((unsigned long long)(uintptr_t)sbase)
               : "memory");
}

// ---------------------------------------------------------------------------
// WMMA helpers (CDNA5 / gfx1250, wave32).
// D(16x16,f32) = A(16x32,f16) * B(32x16,f16) + C
// A frag layout: lane L -> row m = L%16, g = L/16; vgpr j holds half-pair at
//   k = g*8 + 2j (j<4)  or  k = 16 + g*8 + 2(j-4) (j>=4)
// B frag layout: lane L -> col n = L%16, g = L/16; vgpr j holds half-pair at
//   k = g*8 + 2j (j<4)  or  k = 16 + g*8 + 2(j-4) (j>=4)
// C/D layout: vgpr r, lanes 0-15 -> (m=r, n=lane); lanes 16-31 -> (m=r+8).
// B matrices are stored transposed ([n][k]) so pairs along k are contiguous.
// ---------------------------------------------------------------------------
static __device__ __forceinline__ v8f wmma32(v16h a, v16h b, v8f c) {
  return __builtin_amdgcn_wmma_f32_16x16x32_f16(false, a, false, b,
                                                (short)0, c, false, false);
}

static __device__ __forceinline__ v16h load_a(const half_t* At, int lda,
                                              int kbase, int lane) {
  const int g = lane >> 4, m = lane & 15;
  const half_t* p = At + (size_t)m * lda + kbase + g * 8;
  union { unsigned u[8]; v16h v; } r;
#pragma unroll
  for (int j = 0; j < 4; ++j) r.u[j] = *(const unsigned*)(p + 2 * j);
#pragma unroll
  for (int j = 0; j < 4; ++j) r.u[4 + j] = *(const unsigned*)(p + 16 + 2 * j);
  return r.v;
}

static __device__ __forceinline__ v16h load_b(const half_t* Bt, int ldb,
                                              int kbase, int lane) {
  const int g = lane >> 4, n = lane & 15;
  const half_t* p = Bt + (size_t)n * ldb + kbase + g * 8;
  union { unsigned u[8]; v16h v; } r;
#pragma unroll
  for (int j = 0; j < 4; ++j) r.u[j] = *(const unsigned*)(p + 2 * j);
#pragma unroll
  for (int j = 0; j < 4; ++j) r.u[4 + j] = *(const unsigned*)(p + 16 + 2 * j);
  return r.v;
}

// ---------------------------------------------------------------------------
// Generic 16x16-tile GEMM: C = act((A * B^T + bias) * alpha)
// One wave per output tile; grid = (rows/16, cols/16).
// Optional outputs: f32 row-major, f16 row-major, f16 transposed ([n][m]).
// ---------------------------------------------------------------------------
__global__ void __launch_bounds__(32) gemm16(
    const half_t* __restrict__ A, int lda,
    const half_t* __restrict__ Bt, int ldb,
    const float* __restrict__ bias, float alpha, int relu,
    float* __restrict__ Cf, int ldcf,
    half_t* __restrict__ Ch, int ldch,
    half_t* __restrict__ Ct, int ldct,
    int Kdim) {
  const int lane = threadIdx.x & 31;
  const int g = lane >> 4, ln = lane & 15;
  const int rowbase = blockIdx.x * 16;
  const int colbase = blockIdx.y * 16;
  const half_t* At = A + (size_t)rowbase * lda;
  const half_t* Bn = Bt + (size_t)colbase * ldb;

  v8f c = {};
  for (int k = 0; k < Kdim; k += 32)
    c = wmma32(load_a(At, lda, k, lane), load_b(Bn, ldb, k, lane), c);

  const float bv = bias ? bias[colbase + ln] : 0.0f;
  float o[8];
#pragma unroll
  for (int r = 0; r < 8; ++r) {
    float v = (c[r] + bv) * alpha;
    if (relu) v = fmaxf(v, 0.0f);
    o[r] = v;
  }
  const int m0 = rowbase + (g << 3);
  if (Cf) {
#pragma unroll
    for (int r = 0; r < 8; ++r)
      Cf[(size_t)(m0 + r) * ldcf + colbase + ln] = o[r];
  }
  if (Ch) {
#pragma unroll
    for (int r = 0; r < 8; ++r)
      Ch[(size_t)(m0 + r) * ldch + colbase + ln] = (half_t)o[r];
  }
  if (Ct) {
    half_t* d = Ct + (size_t)(colbase + ln) * ldct + m0;
#pragma unroll
    for (int r = 0; r < 8; ++r) d[r] = (half_t)o[r];
  }
}

// ---------------------------------------------------------------------------
// Flash attention: one wave per (16-query tile, head). Online softmax, P tile
// is bounced through LDS to convert C-layout -> A-layout for the P*V WMMA.
// q is pre-scaled by 1/sqrt(64).
// ---------------------------------------------------------------------------
__global__ void __launch_bounds__(32) flash_attn(
    const half_t* __restrict__ qh, const half_t* __restrict__ kh,
    const half_t* __restrict__ vT, half_t* __restrict__ mhoT) {
  __shared__ half_t Plds[16 * 32];
  const int lane = threadIdx.x & 31;
  const int g = lane >> 4, ln = lane & 15;
  const int qt = blockIdx.x;
  const int h  = blockIdx.y;
  const half_t* Q  = qh + ((size_t)h * N_TOK + qt * 16) * K_HD;
  const half_t* Kb = kh + (size_t)h * N_TOK * K_HD;
  const half_t* Vt = vT + (size_t)h * K_HD * N_TOK;

  const v16h aq0 = load_a(Q, K_HD, 0, lane);
  const v16h aq1 = load_a(Q, K_HD, 32, lane);

  v8f acc[4];
  float mrow[8], lrow[8];
#pragma unroll
  for (int t = 0; t < 4; ++t) acc[t] = (v8f){};
#pragma unroll
  for (int r = 0; r < 8; ++r) { mrow[r] = -3.0e38f; lrow[r] = 0.0f; }

  for (int jb = 0; jb < N_TOK; jb += 32) {
    // S tile [16 x 32] = q @ k^T  (Kdim = 64 features -> two wmma per n-tile)
    v8f s0 = {}, s1 = {};
    s0 = wmma32(aq0, load_b(Kb + (size_t)jb * K_HD, K_HD, 0, lane), s0);
    s0 = wmma32(aq1, load_b(Kb + (size_t)jb * K_HD, K_HD, 32, lane), s0);
    s1 = wmma32(aq0, load_b(Kb + (size_t)(jb + 16) * K_HD, K_HD, 0, lane), s1);
    s1 = wmma32(aq1, load_b(Kb + (size_t)(jb + 16) * K_HD, K_HD, 32, lane), s1);

    // online softmax per row (rows live in vgpr slot r, per 16-lane half)
    float scale[8];
#pragma unroll
    for (int r = 0; r < 8; ++r) {
      float mx = fmaxf(s0[r], s1[r]);
#pragma unroll
      for (int d = 1; d < 16; d <<= 1) mx = fmaxf(mx, __shfl_xor(mx, d, 32));
      const float mnew = fmaxf(mrow[r], mx);
      const float p0 = __expf(s0[r] - mnew);
      const float p1 = __expf(s1[r] - mnew);
      const float al = __expf(mrow[r] - mnew);
      float sum = p0 + p1;
#pragma unroll
      for (int d = 1; d < 16; d <<= 1) sum += __shfl_xor(sum, d, 32);
      lrow[r] = lrow[r] * al + sum;
      mrow[r] = mnew;
      scale[r] = al;
      s0[r] = p0; s1[r] = p1;
    }
    // C-layout -> row-major f16 in LDS, then re-read as A fragment
    const int m0 = g << 3;
#pragma unroll
    for (int r = 0; r < 8; ++r) {
      Plds[(m0 + r) * 32 + ln]      = (half_t)s0[r];
      Plds[(m0 + r) * 32 + 16 + ln] = (half_t)s1[r];
    }
    const v16h ap = load_a(Plds, 32, 0, lane);  // same-wave LDS ops: in order

#pragma unroll
    for (int t = 0; t < 4; ++t) {
#pragma unroll
      for (int r = 0; r < 8; ++r) acc[t][r] *= scale[r];
    }
    acc[0] = wmma32(ap, load_b(Vt + (size_t)0  * N_TOK + jb, N_TOK, 0, lane), acc[0]);
    acc[1] = wmma32(ap, load_b(Vt + (size_t)16 * N_TOK + jb, N_TOK, 0, lane), acc[1]);
    acc[2] = wmma32(ap, load_b(Vt + (size_t)32 * N_TOK + jb, N_TOK, 0, lane), acc[2]);
    acc[3] = wmma32(ap, load_b(Vt + (size_t)48 * N_TOK + jb, N_TOK, 0, lane), acc[3]);
  }

  // normalize and store transposed head-concat output mhoT[h*64+v][tok]
  const int tok0 = qt * 16 + (g << 3);
#pragma unroll
  for (int t = 0; t < 4; ++t) {
    half_t* d = mhoT + (size_t)(h * 64 + t * 16 + ln) * N_TOK + tok0;
#pragma unroll
    for (int r = 0; r < 8; ++r) d[r] = (half_t)(acc[t][r] / lrow[r]);
  }
}

// ---------------------------------------------------------------------------
// Fused rpe-combine + GEMM:
//   adjusted[16 rows x 256] = (sum_f rpe_w[f]*rpe[f] + rpe_b) @ mho
// One 256-thread block per 16-row stripe. The 335 MB rpe stream (the HBM
// roofline of the whole layer) is staged with double-buffered
// global_load_async_to_lds_b64 DMA: each wave issues next-tile copies before
// combining/WMMA-ing the current tile and retires them with s_wait_asynccnt
// (async loads complete in order, so waiting to <=5 drains exactly the
// current tile while the next 5 stay in flight). Each thread DMAs and
// combines only its own 8-byte slice per channel, so the Raw buffer needs no
// barrier; the single barrier per iteration covers the combined f16 A tile.
// ---------------------------------------------------------------------------
__global__ void __launch_bounds__(256) rpe_gemm(
    const float* __restrict__ rpe, const float* __restrict__ rpe_w,
    const float* __restrict__ rpe_b, const half_t* __restrict__ mhoT,
    half_t* __restrict__ adjF16) {
  __shared__ float  Raw[2][5][512];
  __shared__ half_t Wl[2][16 * 32];
  const int tid  = threadIdx.x;
  const int lane = tid & 31;
  const int wave = tid >> 5;
  const int g = lane >> 4, ln = lane & 15;
  const int rowbase = blockIdx.x * 16;

  float wf[5];
#pragma unroll
  for (int f = 0; f < 5; ++f) wf[f] = rpe_w[f];
  const float bb = rpe_b[0];

  // this thread's 2-element slice of the 16x32 tile
  const int e  = tid * 2;
  const int em = e >> 5;
  const int et = e & 31;
  const unsigned base_off = ((unsigned)(rowbase + em) * N_TOK + et) * 4u;

  const int t0 = wave * 2, t1 = wave * 2 + 1;  // 8 waves x 2 tiles = 256 cols
  const half_t* B0 = mhoT + (size_t)(t0 * 16) * N_TOK;
  const half_t* B1 = mhoT + (size_t)(t1 * 16) * N_TOK;

  // prologue: DMA tile 0 into buffer 0
  {
    const unsigned goff = base_off;
#pragma unroll
    for (int f = 0; f < 5; ++f)
      async_copy_b64(lds_addr(&Raw[0][f][e]), rpe,
                     goff + (unsigned)f * (unsigned)(N_TOK * N_TOK * 4u));
  }

  v8f c0 = {}, c1 = {};
  for (int ks = 0; ks < N_TOK / 32; ++ks) {
    const int buf = ks & 1;
    if (ks + 1 < N_TOK / 32) {
      const unsigned goff = base_off + (unsigned)((ks + 1) * 32) * 4u;
#pragma unroll
      for (int f = 0; f < 5; ++f)
        async_copy_b64(lds_addr(&Raw[buf ^ 1][f][e]), rpe,
                       goff + (unsigned)f * (unsigned)(N_TOK * N_TOK * 4u));
      WAIT_ASYNC(5);  // current tile landed; next tile still in flight
    } else {
      WAIT_ASYNC(0);
    }

    // combine the 5 rpe channels for this thread's own slice (no barrier
    // needed: producer == consumer for Raw)
    float a0 = bb, a1 = bb;
#pragma unroll
    for (int f = 0; f < 5; ++f) {
      a0 = fmaf(wf[f], Raw[buf][f][e], a0);
      a1 = fmaf(wf[f], Raw[buf][f][e + 1], a1);
    }
    Wl[buf][e]     = (half_t)a0;
    Wl[buf][e + 1] = (half_t)a1;
    __syncthreads();  // A tile visible to all 8 waves

    const v16h a = load_a(Wl[buf], 32, 0, lane);
    const int jb = ks * 32;
    c0 = wmma32(a, load_b(B0 + jb, N_TOK, 0, lane), c0);
    c1 = wmma32(a, load_b(B1 + jb, N_TOK, 0, lane), c1);
  }

  const int m0 = rowbase + (g << 3);
#pragma unroll
  for (int r = 0; r < 8; ++r) {
    adjF16[(size_t)(m0 + r) * HV + t0 * 16 + ln] = (half_t)c0[r];
    adjF16[(size_t)(m0 + r) * HV + t1 * 16 + ln] = (half_t)c1[r];
  }
}

// ---------------------------------------------------------------------------
// out = LayerNorm(a + b); optional f16 copy of the result.
// One 128-thread block per row (D = 128).
// ---------------------------------------------------------------------------
__global__ void __launch_bounds__(128) add_ln(
    const float* __restrict__ a, const float* __restrict__ b,
    const float* __restrict__ gam, const float* __restrict__ bet,
    float* __restrict__ outf, half_t* __restrict__ outh) {
  __shared__ float red[4];
  const int row = blockIdx.x;
  const int t = threadIdx.x;
  const size_t idx = (size_t)row * D_MOD + t;
  const float v = a[idx] + b[idx];

  float s = v;
#pragma unroll
  for (int d = 1; d < 32; d <<= 1) s += __shfl_xor(s, d, 32);
  if ((t & 31) == 0) red[t >> 5] = s;
  __syncthreads();
  const float mean = (red[0] + red[1] + red[2] + red[3]) * (1.0f / D_MOD);
  __syncthreads();

  const float dv = v - mean;
  float s2 = dv * dv;
#pragma unroll
  for (int d = 1; d < 32; d <<= 1) s2 += __shfl_xor(s2, d, 32);
  if ((t & 31) == 0) red[t >> 5] = s2;
  __syncthreads();
  const float var = (red[0] + red[1] + red[2] + red[3]) * (1.0f / D_MOD);

  const float y = dv * rsqrtf(var + 1e-5f) * gam[t] + bet[t];
  outf[idx] = y;
  if (outh) outh[idx] = (half_t)y;
}

// ---------------------------------------------------------------------------
// Prep: f32 -> f16 copy, and batched f32 [R][C] -> f16 [C][R] transpose.
// ---------------------------------------------------------------------------
__global__ void cvt_f16(const float* __restrict__ src, half_t* __restrict__ dst,
                        int n) {
  const int i = blockIdx.x * blockDim.x + threadIdx.x;
  if (i < n) dst[i] = (half_t)src[i];
}

__global__ void tr_f16(const float* __restrict__ src, half_t* __restrict__ dst,
                       int R, int C) {
  const size_t base = (size_t)blockIdx.y * R * C;
  const int i = blockIdx.x * blockDim.x + threadIdx.x;
  if (i < R * C) {
    const int r = i / C, c = i % C;
    dst[base + (size_t)c * R + r] = (half_t)src[base + i];
  }
}

// ---------------------------------------------------------------------------
extern "C" void kernel_launch(void* const* d_in, const int* in_sizes, int n_in,
                              void* d_out, int out_size, void* d_ws,
                              size_t ws_size, hipStream_t stream) {
  (void)in_sizes; (void)n_in; (void)out_size; (void)ws_size;
  const float* x     = (const float*)d_in[0];
  const float* rpe   = (const float*)d_in[1];
  const float* Wq    = (const float*)d_in[2];
  const float* bq    = (const float*)d_in[3];
  const float* Wk    = (const float*)d_in[4];
  const float* bk    = (const float*)d_in[5];
  const float* Wv    = (const float*)d_in[6];
  const float* bv    = (const float*)d_in[7];
  const float* rpe_w = (const float*)d_in[8];
  const float* rpe_b = (const float*)d_in[9];
  const float* Wo    = (const float*)d_in[10];
  const float* bo    = (const float*)d_in[11];
  const float* W1    = (const float*)d_in[12];
  const float* b1    = (const float*)d_in[13];
  const float* W2    = (const float*)d_in[14];
  const float* b2    = (const float*)d_in[15];
  const float* ln1g  = (const float*)d_in[16];
  const float* ln1b  = (const float*)d_in[17];
  const float* ln2g  = (const float*)d_in[18];
  const float* ln2b  = (const float*)d_in[19];
  float* out = (float*)d_out;

  char* w = (char*)d_ws;
  auto alloc = [&](size_t bytes) {
    char* p = w;
    w += (bytes + 255) & ~(size_t)255;
    return p;
  };
  half_t* xh    = (half_t*)alloc((size_t)N_TOK * D_MOD * 2);
  half_t* qh    = (half_t*)alloc((size_t)H_N * N_TOK * K_HD * 2);
  half_t* kh    = (half_t*)alloc((size_t)H_N * N_TOK * K_HD * 2);
  half_t* vT    = (half_t*)alloc((size_t)H_N * K_HD * N_TOK * 2);
  half_t* mhoT  = (half_t*)alloc((size_t)HV * N_TOK * 2);
  half_t* adjF  = (half_t*)alloc((size_t)N_TOK * HV * 2);
  half_t* x1h   = (half_t*)alloc((size_t)N_TOK * D_MOD * 2);
  half_t* hff   = (half_t*)alloc((size_t)N_TOK * F_FFN * 2);
  float*  projf = (float*)alloc((size_t)N_TOK * D_MOD * 4);
  float*  ffn2f = (float*)alloc((size_t)N_TOK * D_MOD * 4);
  float*  x1f   = (float*)alloc((size_t)N_TOK * D_MOD * 4);
  half_t* WqT   = (half_t*)alloc((size_t)H_N * K_HD * D_MOD * 2);
  half_t* WkT   = (half_t*)alloc((size_t)H_N * K_HD * D_MOD * 2);
  half_t* WvT   = (half_t*)alloc((size_t)H_N * K_HD * D_MOD * 2);
  half_t* WoT   = (half_t*)alloc((size_t)D_MOD * HV * 2);
  half_t* W1T   = (half_t*)alloc((size_t)F_FFN * D_MOD * 2);
  half_t* W2T   = (half_t*)alloc((size_t)D_MOD * F_FFN * 2);

  // ---- prep: f16 conversion + weight transposes (B operands are [n][k]) ----
  cvt_f16<<<dim3((N_TOK * D_MOD + 255) / 256), 256, 0, stream>>>(
      x, xh, N_TOK * D_MOD);
  tr_f16<<<dim3((D_MOD * K_HD + 255) / 256, H_N), 256, 0, stream>>>(
      Wq, WqT, D_MOD, K_HD);
  tr_f16<<<dim3((D_MOD * K_HD + 255) / 256, H_N), 256, 0, stream>>>(
      Wk, WkT, D_MOD, K_HD);
  tr_f16<<<dim3((D_MOD * K_HD + 255) / 256, H_N), 256, 0, stream>>>(
      Wv, WvT, D_MOD, K_HD);
  tr_f16<<<dim3((HV * D_MOD + 255) / 256, 1), 256, 0, stream>>>(
      Wo, WoT, HV, D_MOD);
  tr_f16<<<dim3((D_MOD * F_FFN + 255) / 256, 1), 256, 0, stream>>>(
      W1, W1T, D_MOD, F_FFN);
  tr_f16<<<dim3((F_FFN * D_MOD + 255) / 256, 1), 256, 0, stream>>>(
      W2, W2T, F_FFN, D_MOD);

  // ---- per-head q/k/v projections (q carries the 1/sqrt(K)=0.125 scale) ----
  const dim3 gqkv(N_TOK / 16, K_HD / 16);
  for (int h = 0; h < H_N; ++h) {
    gemm16<<<gqkv, 32, 0, stream>>>(
        xh, D_MOD, WqT + (size_t)h * K_HD * D_MOD, D_MOD, bq + h * K_HD,
        0.125f, 1, nullptr, 0, qh + (size_t)h * N_TOK * K_HD, K_HD,
        nullptr, 0, D_MOD);
    gemm16<<<gqkv, 32, 0, stream>>>(
        xh, D_MOD, WkT + (size_t)h * K_HD * D_MOD, D_MOD, bk + h * K_HD,
        1.0f, 1, nullptr, 0, kh + (size_t)h * N_TOK * K_HD, K_HD,
        nullptr, 0, D_MOD);
    gemm16<<<gqkv, 32, 0, stream>>>(
        xh, D_MOD, WvT + (size_t)h * K_HD * D_MOD, D_MOD, bv + h * K_HD,
        1.0f, 1, nullptr, 0, nullptr, 0,
        vT + (size_t)h * K_HD * N_TOK, N_TOK, D_MOD);
  }

  // ---- flash attention -> mhoT [256][4096] f16 ----
  flash_attn<<<dim3(N_TOK / 16, H_N), 32, 0, stream>>>(qh, kh, vT, mhoT);

  // ---- fused rpe combine + (rpe_weights @ mho) -> adjF16 [4096][256] ----
  rpe_gemm<<<dim3(N_TOK / 16), 256, 0, stream>>>(rpe, rpe_w, rpe_b, mhoT, adjF);

  // ---- proj = adjusted @ Wo + bo -> f32 ----
  gemm16<<<dim3(N_TOK / 16, D_MOD / 16), 32, 0, stream>>>(
      adjF, HV, WoT, HV, bo, 1.0f, 0, projf, D_MOD,
      nullptr, 0, nullptr, 0, HV);

  // ---- x1 = LN1(x + proj) (f32 + f16 copies) ----
  add_ln<<<N_TOK, 128, 0, stream>>>(x, projf, ln1g, ln1b, x1f, x1h);

  // ---- FFN ----
  gemm16<<<dim3(N_TOK / 16, F_FFN / 16), 32, 0, stream>>>(
      x1h, D_MOD, W1T, D_MOD, b1, 1.0f, 1, nullptr, 0, hff, F_FFN,
      nullptr, 0, D_MOD);
  gemm16<<<dim3(N_TOK / 16, D_MOD / 16), 32, 0, stream>>>(
      hff, F_FFN, W2T, F_FFN, b2, 1.0f, 0, ffn2f, D_MOD,
      nullptr, 0, nullptr, 0, F_FFN);

  // ---- out = LN2(x1 + ffn) ----
  add_ln<<<N_TOK, 128, 0, stream>>>(x1f, ffn2f, ln2g, ln2b, out, nullptr);
}
